// SelfParallelRetention_32727650795908
// MI455X (gfx1250) — matrix operand order repaired and checked by reading, not compiled, not run on hardware
//
#include <hip/hip_runtime.h>

typedef _Float16 v8h  __attribute__((ext_vector_type(8)));
typedef _Float16 v16h __attribute__((ext_vector_type(16)));
typedef float    v8f  __attribute__((ext_vector_type(8)));

#define NH 12
#define SEQ 50
#define DH 64
#define EMB 768
#define SP 64            // seq padded to 64 (4 x 16 M-tiles)
#define XS 776           // x LDS row stride in halfs (768 + 8 pad, keeps 16B align)
#define QS 72            // q/k/vT/ret LDS row stride in halfs (64 + 8 pad)
#define NTHREADS 384     // 12 waves (wave32)
#define MTOT 1024

#define SMEM_HALFS (SP * XS + 4 * SP * QS)
#define SMEM_BYTES (SMEM_HALFS * 2 + 8)

#define W16_HALFS  (3 * EMB * EMB)             // f16 weights in ws
#define X16_OFFSET W16_HALFS                   // f16 x follows (offset in halfs, 16B aligned)

#define WMMA_F16(a, b, c) \
  __builtin_amdgcn_wmma_f32_16x16x32_f16(false, (a), false, (b), (short)0, (c), false, false)

// A operand (16x32 f16): lane row = lane&15; K = {hi*8..+7, 16+hi*8..+7}
__device__ __forceinline__ v16h frag_a(const _Float16* p, int hi) {
  v8h lo = *(const v8h*)(p + hi * 8);
  v8h h8 = *(const v8h*)(p + 16 + hi * 8);
  return __builtin_shufflevector(lo, h8, 0, 1, 2, 3, 4, 5, 6, 7, 8, 9, 10, 11, 12, 13, 14, 15);
}
// B operand (32x16 f16): lane col = lane&15; K = {hi*8..+7, 16+hi*8..+7}
__device__ __forceinline__ v16h frag_b(const _Float16* p, int hi) {
  v8h lo = *(const v8h*)(p + hi * 8);
  v8h h8 = *(const v8h*)(p + 16 + hi * 8);
  return __builtin_shufflevector(lo, h8, 0, 1, 2, 3, 4, 5, 6, 7, 8, 9, 10, 11, 12, 13, 14, 15);
}

// ---- prep 1: convert the three 768x768 fp32 weight matrices to f16 in workspace ----
__global__ void cvt_weights(const float* __restrict__ wq, const float* __restrict__ wk,
                            const float* __restrict__ wv, _Float16* __restrict__ o) {
  const int N = EMB * EMB;
  int i = blockIdx.x * blockDim.x + threadIdx.x;
  if (i < N)            o[i] = (_Float16)wq[i];
  else if (i < 2 * N)   o[i] = (_Float16)wk[i - N];
  else if (i < 3 * N)   o[i] = (_Float16)wv[i - 2 * N];
}

// ---- prep 2: convert x (m,s,e) fp32 -> f16 once; read 157MB, write 78.6MB ----
__global__ void cvt_x(const float* __restrict__ x, _Float16* __restrict__ o, int n4) {
  int i = blockIdx.x * blockDim.x + threadIdx.x;
  if (i < n4) {
    float4 v = ((const float4*)x)[i];
    _Float16* p = o + (size_t)i * 4;
    p[0] = (_Float16)v.x; p[1] = (_Float16)v.y;
    p[2] = (_Float16)v.z; p[3] = (_Float16)v.w;
  }
}

// ---- fused retention: one workgroup per (m, h) ----
__global__ __launch_bounds__(NTHREADS, 1) void retention_kernel(
    const _Float16* __restrict__ x16, const float* __restrict__ bq,
    const float* __restrict__ bk, const float* __restrict__ bv,
    const _Float16* __restrict__ w16, float* __restrict__ out) {
  extern __shared__ __align__(16) unsigned char smem_raw[];
  _Float16* xlds  = (_Float16*)smem_raw;          // SP x XS  (x[m] f16, rows >= 50 zero)
  _Float16* qlds  = xlds + SP * XS;               // SP x QS  row-major (s, d)
  _Float16* klds  = qlds + SP * QS;               // SP x QS  row-major (s, d)
  _Float16* vtlds = klds + SP * QS;               // SP x QS  TRANSPOSED: (d, s)
  _Float16* retld = vtlds + SP * QS;              // SP x QS  row-major (s, t)
  float*    red   = (float*)(retld + SP * QS);    // [0]=sum  [1]=sumsq

  const int tid  = threadIdx.x;
  const int wave = tid >> 5;
  const int lane = tid & 31;
  const int lo   = lane & 15;
  const int hi   = lane >> 4;
  const int bid  = blockIdx.x;
  const int mIdx = bid / NH;
  const int h    = bid - mIdx * NH;

  if (tid == 0) { red[0] = 0.0f; red[1] = 0.0f; }

  // ---- stage x[m] into LDS as 16B vectors, zero-pad rows 50..63 ----
  const _Float16* xg = x16 + (size_t)mIdx * SEQ * EMB;
  const v8h zero8 = {};
  for (int i = tid; i < SP * (EMB / 8); i += NTHREADS) {   // 6144 vectors, 16 per thread
    int r  = i / (EMB / 8);
    int c8 = i - r * (EMB / 8);
    v8h v = (r < SEQ) ? *(const v8h*)(xg + r * EMB + c8 * 8) : zero8;
    *(v8h*)&xlds[r * XS + c8 * 8] = v;
  }
  __syncthreads();

  // ---- phase 1: q/k/v projections. wave -> (proj, n-tile); B reused over 4 M-tiles ----
  {
    const int p  = wave >> 2;          // 0=q 1=k 2=v
    const int nt = wave & 3;
    const int f0 = h * DH + nt * 16;   // output feature base
    const _Float16* wrow = w16 + (size_t)p * EMB * EMB + (size_t)(f0 + lo) * EMB;
    const float* bias_p = (p == 0) ? bq : (p == 1) ? bk : bv;
    const _Float16* arow = &xlds[lo * XS];
    v8f acc[4] = {};
    for (int kk = 0; kk < EMB; kk += 32) {
      v16h bf = frag_b(wrow + kk, hi);
      // distinct fragments -> one ds_load clause, then 4 WMMAs back-to-back
      v16h a0 = frag_a(arow + 0 * 16 * XS + kk, hi);
      v16h a1 = frag_a(arow + 1 * 16 * XS + kk, hi);
      v16h a2 = frag_a(arow + 2 * 16 * XS + kk, hi);
      v16h a3 = frag_a(arow + 3 * 16 * XS + kk, hi);
      acc[0] = WMMA_F16(a0, bf, acc[0]);
      acc[1] = WMMA_F16(a1, bf, acc[1]);
      acc[2] = WMMA_F16(a2, bf, acc[2]);
      acc[3] = WMMA_F16(a3, bf, acc[3]);
    }
    float bias = bias_p[f0 + lo];
#pragma unroll
    for (int mt = 0; mt < 4; ++mt) {
#pragma unroll
      for (int r = 0; r < 8; ++r) {
        int row = mt * 16 + hi * 8 + r;           // s index
        int col = nt * 16 + lo;                   // d index (within head)
        float val = (row < SEQ) ? (acc[mt][r] + bias) : 0.0f;  // keep padding exactly 0
        if (p == 0)      qlds[row * QS + col] = (_Float16)val;
        else if (p == 1) klds[row * QS + col] = (_Float16)val;
        else             vtlds[col * QS + row] = (_Float16)val;  // store v transposed
      }
    }
  }
  __syncthreads();

  // ---- phase 2: retention = (q @ k^T) * decay ----
  {
    const int nTiles = (wave < 4) ? 2 : 1;
    for (int j = 0; j < nTiles; ++j) {
      int t = wave + j * 12;
      int mt = t >> 2, nt = t & 3;
      v8f acc = {};
#pragma unroll
      for (int kk = 0; kk < DH; kk += 32) {
        v16h af = frag_a(&qlds[(mt * 16 + lo) * QS + kk], hi);   // A = q (s x d)
        v16h bf = frag_b(&klds[(nt * 16 + lo) * QS + kk], hi);   // B = k^T (col n = k row)
        acc = WMMA_F16(af, bf, acc);
      }
#pragma unroll
      for (int r = 0; r < 8; ++r) {
        int srow = mt * 16 + hi * 8 + r;
        int tcol = nt * 16 + lo;
        float dec = 1.0f / (1.0f + fabsf((float)(srow - tcol)));
        retld[srow * QS + tcol] = (_Float16)(acc[r] * dec);
      }
    }
  }
  __syncthreads();

  // ---- phase 3: out = retention @ v + k; group-norm stats on the fly ----
  v8f oacc[2];
  const int nOut = (wave < 4) ? 2 : 1;
  float sum = 0.0f, sumsq = 0.0f;
  for (int j = 0; j < nOut; ++j) {
    int t = wave + j * 12;
    int mt = t >> 2, nt = t & 3;
    v8f acc = {};
#pragma unroll
    for (int kk = 0; kk < SP; kk += 32) {
      v16h af = frag_a(&retld[(mt * 16 + lo) * QS + kk], hi);    // A = retention (s x t)
      v16h bf = frag_b(&vtlds[(nt * 16 + lo) * QS + kk], hi);    // B = v (t x d) via v^T
      acc = WMMA_F16(af, bf, acc);
    }
#pragma unroll
    for (int r = 0; r < 8; ++r) {
      int srow = mt * 16 + hi * 8 + r;
      int dcol = nt * 16 + lo;
      float val = acc[r] + (float)klds[srow * QS + dcol];        // "+ k" from reference
      acc[r] = val;
      sum += val;
      sumsq += val * val;
    }
    oacc[j] = acc;
  }
  // wave reduce then LDS atomic (padded rows contribute exact zeros)
#pragma unroll
  for (int off = 16; off > 0; off >>= 1) {
    sum += __shfl_xor(sum, off, 32);
    sumsq += __shfl_xor(sumsq, off, 32);
  }
  if (lane == 0) {
    atomicAdd(&red[0], sum);
    atomicAdd(&red[1], sumsq);
  }
  __syncthreads();

  const float invN = 1.0f / (float)(SEQ * DH);   // 3200 valid elements
  float mean = red[0] * invN;
  float var  = red[1] * invN - mean * mean;
  float rstd = rsqrtf(var + 1e-5f);

  float* op = out + (size_t)bid * SEQ * DH;      // raw (m, H, s, d) flatten
  for (int j = 0; j < nOut; ++j) {
    int t = wave + j * 12;
    int mt = t >> 2, nt = t & 3;
#pragma unroll
    for (int r = 0; r < 8; ++r) {
      int srow = mt * 16 + hi * 8 + r;
      if (srow < SEQ) {
        int dcol = nt * 16 + lo;
        op[srow * DH + dcol] = (oacc[j][r] - mean) * rstd;
      }
    }
  }
}

extern "C" void kernel_launch(void* const* d_in, const int* in_sizes, int n_in,
                              void* d_out, int out_size, void* d_ws, size_t ws_size,
                              hipStream_t stream) {
  (void)in_sizes; (void)n_in; (void)out_size; (void)ws_size;
  const float* x  = (const float*)d_in[0];
  const float* Wq = (const float*)d_in[1];
  const float* bq = (const float*)d_in[2];
  const float* Wk = (const float*)d_in[3];
  const float* bk = (const float*)d_in[4];
  const float* Wv = (const float*)d_in[5];
  const float* bv = (const float*)d_in[6];
  _Float16* w16 = (_Float16*)d_ws;               // 3.4 MB
  _Float16* x16 = w16 + X16_OFFSET;              // 78.6 MB (ws total ~82 MB)

  const int nW = 3 * EMB * EMB;
  cvt_weights<<<(nW + 255) / 256, 256, 0, stream>>>(Wq, Wk, Wv, w16);

  const int nX4 = (MTOT * SEQ * EMB) / 4;
  cvt_x<<<(nX4 + 255) / 256, 256, 0, stream>>>(x, x16, nX4);

  static int attr_set = 0;  // idempotent host-side attribute (not a stream op)
  if (!attr_set) {
    hipFuncSetAttribute((const void*)retention_kernel,
                        hipFuncAttributeMaxDynamicSharedMemorySize, SMEM_BYTES);
    attr_set = 1;
  }

  dim3 grid(MTOT * NH);   // one workgroup per (m, h)
  dim3 block(NTHREADS);
  retention_kernel<<<grid, block, SMEM_BYTES, stream>>>(x16, bq, bk, bv, w16, (float*)d_out);
}
